// S3Tokenizer_76733885710491
// MI455X (gfx1250) — matrix operand order repaired and checked by reading, not compiled, not run on hardware
//
#include <hip/hip_runtime.h>
#include <hip/hip_bf16.h>

// ---------------------------------------------------------------------------
// S3 tokenizer on gfx1250 (wave32 WMMA, bf16 x bf16 -> f32).
// Conv encoder as im2col GEMM + codebook argmin as GEMM on
// v_wmma_f32_16x16x32_bf16; conv uses ASYNCcnt global->LDS copies when the
// gfx1250 async builtins are available (reg-staged fallback otherwise).
// ---------------------------------------------------------------------------

typedef __attribute__((ext_vector_type(16))) __bf16 v16bf;
typedef __attribute__((ext_vector_type(8)))  __bf16 v8bf;
typedef __attribute__((ext_vector_type(8)))  float  v8f;
typedef __attribute__((ext_vector_type(4)))  int    v4i;

#define BF16_FRAG(lo, hi) \
  __builtin_shufflevector((lo), (hi), 0,1,2,3,4,5,6,7,8,9,10,11,12,13,14,15)

#define WMMA_BF16(A, B, C) \
  __builtin_amdgcn_wmma_f32_16x16x32_bf16(false, (A), false, (B), (short)0, (C), false, false)

#if defined(__has_builtin)
#if __has_builtin(__builtin_amdgcn_global_load_async_to_lds_b128) && \
    __has_builtin(__builtin_amdgcn_s_wait_asynccnt)
#define USE_ASYNC_LDS 1
#endif
#endif
#ifndef USE_ASYNC_LDS
#define USE_ASYNC_LDS 0
#endif

#if USE_ASYNC_LDS
typedef __attribute__((address_space(1))) v4i as1_v4i;  // global (64-bit ptr)
typedef __attribute__((address_space(3))) v4i as3_v4i;  // LDS (32-bit ptr)
__device__ __forceinline__ void async_b128(const void* g, void* l) {
  // flat->AS1 is numerically identity; flat LDS addr low 32 bits = LDS offset.
  __builtin_amdgcn_global_load_async_to_lds_b128(
      (as1_v4i*)(size_t)g, (as3_v4i*)(unsigned)(size_t)l, 0, 0);
}
#endif

__device__ __forceinline__ __bf16 f2bf(float x) {
  union { float f; unsigned u; } v; v.f = x;
  unsigned r = v.u + 0x7FFFu + ((v.u >> 16) & 1u);   // round-to-nearest-even
  unsigned short hs = (unsigned short)(r >> 16);
  __bf16 out; __builtin_memcpy(&out, &hs, 2); return out;
}

// ---------------------------------------------------------------------------
// Prep kernels (layout conversion to bf16)
// ---------------------------------------------------------------------------

// mels (B, 128, T) f32 -> X0 (B, T, 128) bf16
__global__ void prep_mels_kernel(const float* __restrict__ mels,
                                 __bf16* __restrict__ X0, int T) {
  int t = blockIdx.x, b = blockIdx.y, c = threadIdx.x;
  X0[((size_t)b * T + t) * 128 + c] = f2bf(mels[((size_t)b * 128 + c) * T + t]);
}

// w (C_out, C_in, 3) f32 -> wp (C_out, 3, C_in) bf16 (tap-major K dim)
__global__ void prep_w_kernel(const float* __restrict__ w,
                              __bf16* __restrict__ wp, int C_out, int C_in) {
  int i = blockIdx.x * 256 + threadIdx.x;
  int total = C_out * 3 * C_in;
  if (i >= total) return;
  int co  = i / (3 * C_in);
  int rem = i - co * 3 * C_in;
  int tap = rem / C_in;
  int ci  = rem - tap * C_in;
  wp[i] = f2bf(w[((size_t)co * C_in + ci) * 3 + tap]);
}

// codebook (N, 1024) f32 -> bf16 + per-code squared norm
__global__ __launch_bounds__(256) void prep_codebook_kernel(
    const float* __restrict__ cb, __bf16* __restrict__ cbb,
    float* __restrict__ c2) {
  __shared__ float red[256];
  int c = blockIdx.x, tid = threadIdx.x;
  float s = 0.f;
  for (int i = tid; i < 1024; i += 256) {
    float v = cb[(size_t)c * 1024 + i];
    cbb[(size_t)c * 1024 + i] = f2bf(v);
    s += v * v;
  }
  red[tid] = s; __syncthreads();
  for (int off = 128; off; off >>= 1) {
    if (tid < off) red[tid] += red[tid + off];
    __syncthreads();
  }
  if (tid == 0) c2[c] = red[0];
}

__global__ void zrow_kernel(__bf16* __restrict__ zr) {
  zr[threadIdx.x] = f2bf(0.0f);          // 64-element zero row (pad redirect)
}

__global__ void lens_kernel(const int* __restrict__ ml,
                            long long* __restrict__ out, int n_tok_rows) {
  int i = threadIdx.x;
  if (i < 8) out[n_tok_rows + i] = (long long)(ml[i] >> 2);
}

// ---------------------------------------------------------------------------
// Conv1d (k=3, pad=1, stride s) as WMMA GEMM.
//   X: (B, T_in, C_in) bf16   W: (C_out, 3*C_in) bf16 tap-major
//   Y: (B, T_out, C_out) bf16  ZR: 64 bf16 zeros (OOB-tap redirect)
// Block 256 thr = 8 waves; block tile 64(co) x 128(t); wave tile 32x32.
// Key trick: tap-major K over (T,C) layout makes the im2col source a uniform
// +32-element/stage contiguous stream; only a 3-bit tap-validity mask varies.
// Double-buffered LDS; async global->LDS when available; 1 barrier/stage.
// ---------------------------------------------------------------------------
#define AST 40   // sA row stride (80 B -> 20-bank lane stride, conflict-free)
#define BST 40   // sB row stride

__global__ __launch_bounds__(256) void conv_gemm_kernel(
    const __bf16* __restrict__ X, const __bf16* __restrict__ W,
    const float* __restrict__ bias, __bf16* __restrict__ Y,
    const __bf16* __restrict__ ZR,
    int T_in, int C_in, int T_out, int C_out, int stride, int relu) {
  __shared__ __align__(16) __bf16 sA[2][64 * AST];    // 10 KB
  __shared__ __align__(16) __bf16 sB[2][128 * BST];   // 20 KB

  const int tid  = threadIdx.x;
  const int b    = blockIdx.z;
  const int co0  = blockIdx.y * 64;
  const int t0   = blockIdx.x * 128;
  const int lane = tid & 31;
  const int wv   = tid >> 5;
  const int wm   = wv >> 2;          // 0..1 : co rows wm*32..+31
  const int wn   = wv & 3;           // 0..3 : t cols wn*32..+31
  const int mr   = lane & 15;
  const int h    = lane >> 4;

  // Loader mapping: sA 64x32 -> 8 bf16/thr ; sB 128x32 -> 16 bf16/thr
  const int rowA = tid >> 2, colA = (tid & 3) << 3;
  const int rowB = tid >> 1, colB = (tid & 1) << 4;

  const int nci = C_in >> 5;
  const int nst = 3 * nci;

  // A stream: W rows are K-contiguous -> +32 elements per stage.
  const __bf16* gA = W + (size_t)(co0 + rowA) * (3 * C_in) + colA;
  // B stream: addr = stbase*C_in + colB + 32*stage (uniform across taps).
  const int stbase = stride * (t0 + rowB) - 1;
  const __bf16* gB = X + (size_t)b * T_in * C_in
                       + (long long)stbase * C_in + colB;
  // Per-tap validity (rows beyond T_out only feed unused output columns).
  const int st1 = stbase + 1, st2 = stbase + 2;
  const unsigned vmask = (unsigned)(stbase >= 0 && stbase < T_in)
                       | ((unsigned)(st1 >= 0 && st1 < T_in) << 1)
                       | ((unsigned)(st2 >= 0 && st2 < T_in) << 2);
  int tapI = 0, ciI = 0;                   // next stage to issue/load

  v8f acc00 = {}, acc01 = {}, acc10 = {}, acc11 = {};

#if USE_ASYNC_LDS
#define ISSUE_STAGE(bufi)                                              \
  {                                                                    \
    bool v = (vmask >> tapI) & 1u;                                     \
    const __bf16* srcB = v ? gB : (ZR + colB);                         \
    async_b128(gA,       sA[bufi] + rowA * AST + colA);                \
    async_b128(srcB,     sB[bufi] + rowB * BST + colB);                \
    async_b128(srcB + 8, sB[bufi] + rowB * BST + colB + 8);            \
    gA += 32; gB += 32;                                                \
    if (++ciI == nci) { ciI = 0; ++tapI; }                             \
  }
  ISSUE_STAGE(0);
  for (int s = 0; s < nst; ++s) {
    const int buf = s & 1;
    __builtin_amdgcn_s_wait_asynccnt(0);   // own stage-s copies landed in LDS
    __syncthreads();                       // everyone's landed; prev reads done
    if (s + 1 < nst) ISSUE_STAGE(buf ^ 1); // next stage flies under the WMMAs
#else
  uint4 rA, rB0, rB1;
#define LOAD_STAGE_REGS()                                              \
  {                                                                    \
    bool v = (vmask >> tapI) & 1u;                                     \
    const uint4* g = (const uint4*)(v ? gB : (ZR + colB));             \
    rA = *(const uint4*)gA;                                            \
    rB0 = g[0]; rB1 = g[1];                                            \
    gA += 32; gB += 32;                                                \
    if (++ciI == nci) { ciI = 0; ++tapI; }                             \
  }
  LOAD_STAGE_REGS();
  for (int s = 0; s < nst; ++s) {
    const int buf = s & 1;
    *(uint4*)(sA[buf] + rowA * AST + colA)     = rA;
    *(uint4*)(sB[buf] + rowB * BST + colB)     = rB0;
    *(uint4*)(sB[buf] + rowB * BST + colB + 8) = rB1;
    __syncthreads();
    if (s + 1 < nst) LOAD_STAGE_REGS();
#endif

    const __bf16* ap = sA[buf] + (wm * 32 + mr) * AST;
    v16bf a0 = BF16_FRAG(*(const v8bf*)(ap + h * 16),
                         *(const v8bf*)(ap + h * 16 + 8));
    v16bf a1 = BF16_FRAG(*(const v8bf*)(ap + 16 * AST + h * 16),
                         *(const v8bf*)(ap + 16 * AST + h * 16 + 8));
    const __bf16* bp = sB[buf] + (wn * 32 + mr) * BST + h * 16;
    v16bf b0 = BF16_FRAG(*(const v8bf*)(bp),            *(const v8bf*)(bp + 8));
    v16bf b1 = BF16_FRAG(*(const v8bf*)(bp + 16 * BST), *(const v8bf*)(bp + 16 * BST + 8));

    acc00 = WMMA_BF16(a0, b0, acc00);
    acc01 = WMMA_BF16(a0, b1, acc01);
    acc10 = WMMA_BF16(a1, b0, acc10);
    acc11 = WMMA_BF16(a1, b1, acc11);
  }

  // Epilogue: C/D lane layout -> rows m = h*8+r, column n = lane%16
  const int nl = lane & 15;
#pragma unroll
  for (int ni = 0; ni < 2; ++ni) {
    int t = t0 + wn * 32 + ni * 16 + nl;
    if (t >= T_out) continue;
    size_t base = ((size_t)b * T_out + t) * C_out;
#pragma unroll
    for (int mi = 0; mi < 2; ++mi) {
      v8f a = mi ? (ni ? acc11 : acc10) : (ni ? acc01 : acc00);
      int cbase = co0 + wm * 32 + mi * 16 + h * 8;
      __align__(16) __bf16 o[8];
#pragma unroll
      for (int r = 0; r < 8; ++r) {
        float v = a[r] + bias[cbase + r];
        if (relu) v = v > 0.f ? v : 0.f;
        o[r] = f2bf(v);
      }
      *(uint4*)(Y + base + cbase) = *(const uint4*)o;
    }
  }
}

// ---------------------------------------------------------------------------
// Codebook argmin: argmin_c ||c||^2 - 2 e.c  (row norm & sqrt are monotone).
// Block = 16 enc rows resident in LDS; codebook streamed in 256-code x 32-K
// tiles; 8 waves x 32 codes -> 2 WMMA per wave per K-chunk (shared A-frag).
// ---------------------------------------------------------------------------
#define EST 1032  // sE row stride (1024 + 8): 4-bank lane stride, conflict-free
#define CST 40

__global__ __launch_bounds__(256) void argmin_kernel(
    const __bf16* __restrict__ E, const __bf16* __restrict__ CB,
    const float* __restrict__ c2, long long* __restrict__ out, int ncodes) {
  __shared__ __align__(16) __bf16 sE[16 * EST];     // 33 KB resident enc strip
  __shared__ __align__(16) __bf16 sC[256 * CST];    // 20 KB codebook K-chunk

  const int tid  = threadIdx.x;
  const int row0 = blockIdx.x * 16;
  {
    const uint4* src = (const uint4*)(E + (size_t)row0 * 1024);
    for (int i = tid; i < 2048; i += 256) {
      int r = i >> 7, ce = (i & 127) << 3;
      *(uint4*)(sE + r * EST + ce) = src[i];
    }
  }
  const int lane = tid & 31;
  const int wv   = tid >> 5;
  const int mr   = lane & 15;
  const int h    = lane >> 4;

  float bestv[8]; int besti[8];
#pragma unroll
  for (int r = 0; r < 8; ++r) { bestv[r] = 3.4e38f; besti[r] = 0; }

  const uint4 zz = {0, 0, 0, 0};
  for (int nb = 0; nb < ncodes; nb += 256) {
    const int  code = nb + tid;
    const bool cv   = code < ncodes;
    const uint4* gC = (const uint4*)(CB + (size_t)code * 1024);
    if (cv) __builtin_prefetch(c2 + code, 0, 1);

    uint4 r0 = zz, r1 = zz, r2 = zz, r3 = zz;
    if (cv) { r0 = gC[0]; r1 = gC[1]; r2 = gC[2]; r3 = gC[3]; }

    v8f acc0 = {}, acc1 = {};
    for (int kc = 0; kc < 32; ++kc) {             // K = 1024
      __syncthreads();                            // readers of sC done
      {
        uint4* d = (uint4*)(sC + tid * CST);
        d[0] = r0; d[1] = r1; d[2] = r2; d[3] = r3;
      }
      __syncthreads();
      if (kc + 1 < 32 && cv) {                    // next chunk overlaps WMMAs
        r0 = gC[(kc + 1) * 4 + 0]; r1 = gC[(kc + 1) * 4 + 1];
        r2 = gC[(kc + 1) * 4 + 2]; r3 = gC[(kc + 1) * 4 + 3];
      }
      const __bf16* ap = sE + mr * EST + kc * 32;
      v16bf a = BF16_FRAG(*(const v8bf*)(ap + h * 16),
                          *(const v8bf*)(ap + h * 16 + 8));
      const __bf16* bp = sC + (wv * 32 + mr) * CST + h * 16;
      v16bf b0 = BF16_FRAG(*(const v8bf*)(bp),            *(const v8bf*)(bp + 8));
      v16bf b1 = BF16_FRAG(*(const v8bf*)(bp + 16 * CST), *(const v8bf*)(bp + 16 * CST + 8));
      acc0 = WMMA_BF16(a, b0, acc0);
      acc1 = WMMA_BF16(a, b1, acc1);
    }
#pragma unroll
    for (int ni = 0; ni < 2; ++ni) {
      int c = nb + wv * 32 + ni * 16 + mr;
      if (c < ncodes) {
        float cc = c2[c];
        v8f a = ni ? acc1 : acc0;
#pragma unroll
        for (int r = 0; r < 8; ++r) {
          float d = cc - 2.0f * a[r];
          if (d < bestv[r]) { bestv[r] = d; besti[r] = c; }
        }
      }
    }
  }

  // Final reduction; sE region is dead -> alias it for candidate arrays.
  __syncthreads();
  float* redv = (float*)sE;                       // [16][128] f32
  int*   redi = (int*)((char*)sE + 8192);         // [16][128] i32
#pragma unroll
  for (int r = 0; r < 8; ++r) {
    redv[(h * 8 + r) * 128 + wv * 16 + mr] = bestv[r];
    redi[(h * 8 + r) * 128 + wv * 16 + mr] = besti[r];
  }
  __syncthreads();
  if (tid < 16) {
    float bv = 3.4e38f; int bi = 0x7fffffff;
    for (int j = 0; j < 128; ++j) {
      float v = redv[tid * 128 + j]; int ii = redi[tid * 128 + j];
      if (v < bv || (v == bv && ii < bi)) { bv = v; bi = ii; }
    }
    out[row0 + tid] = (long long)bi;
  }
}

// ---------------------------------------------------------------------------
// Host launcher
// ---------------------------------------------------------------------------
extern "C" void kernel_launch(void* const* d_in, const int* in_sizes, int n_in,
                              void* d_out, int out_size, void* d_ws, size_t ws_size,
                              hipStream_t stream) {
  (void)in_sizes; (void)n_in; (void)out_size;
  const float* mels     = (const float*)d_in[0];
  const int*   mel_lens = (const int*)d_in[1];
  const float* w1 = (const float*)d_in[2]; const float* b1 = (const float*)d_in[3];
  const float* w2 = (const float*)d_in[4]; const float* b2 = (const float*)d_in[5];
  const float* w3 = (const float*)d_in[6]; const float* b3 = (const float*)d_in[7];
  const float* w4 = (const float*)d_in[8]; const float* b4 = (const float*)d_in[9];
  const float* cb = (const float*)d_in[10];
  long long* out = (long long*)d_out;

  const int B = 8, T1 = 4000, T2 = 2000, T3 = 1000, NC = 6561;

  size_t off = 0;
  auto alloc = [&](size_t bytes) {
    size_t o = off; off = (off + bytes + 255) & ~(size_t)255; return o;
  };
  size_t o_buf0 = alloc((size_t)B * T2 * 512 * 2);   // X0 / H2 / H4
  size_t o_buf1 = alloc((size_t)B * T1 * 512 * 2);   // H1 / H3
  size_t o_w1 = alloc((size_t)512  * 3 * 128  * 2);
  size_t o_w2 = alloc((size_t)512  * 3 * 512  * 2);
  size_t o_w3 = alloc((size_t)1024 * 3 * 512  * 2);
  size_t o_w4 = alloc((size_t)1024 * 3 * 1024 * 2);
  size_t o_cb = alloc((size_t)NC * 1024 * 2);
  size_t o_c2 = alloc((size_t)NC * 4);
  size_t o_zr = alloc((size_t)64 * 2);
  if (off > ws_size) return;

  char* ws = (char*)d_ws;
  __bf16* X0  = (__bf16*)(ws + o_buf0);
  __bf16* H1  = (__bf16*)(ws + o_buf1);
  __bf16* H2  = (__bf16*)(ws + o_buf0);
  __bf16* H3  = (__bf16*)(ws + o_buf1);
  __bf16* H4  = (__bf16*)(ws + o_buf0);
  __bf16* WP1 = (__bf16*)(ws + o_w1);
  __bf16* WP2 = (__bf16*)(ws + o_w2);
  __bf16* WP3 = (__bf16*)(ws + o_w3);
  __bf16* WP4 = (__bf16*)(ws + o_w4);
  __bf16* CBB = (__bf16*)(ws + o_cb);
  float*  C2  = (float*)(ws + o_c2);
  __bf16* ZR  = (__bf16*)(ws + o_zr);

  // --- prep ---
  zrow_kernel<<<1, 64, 0, stream>>>(ZR);
  prep_mels_kernel<<<dim3(T1, B), 128, 0, stream>>>(mels, X0, T1);
  prep_w_kernel<<<(512 * 3 * 128 + 255) / 256, 256, 0, stream>>>(w1, WP1, 512, 128);
  prep_w_kernel<<<(512 * 3 * 512 + 255) / 256, 256, 0, stream>>>(w2, WP2, 512, 512);
  prep_w_kernel<<<(1024 * 3 * 512 + 255) / 256, 256, 0, stream>>>(w3, WP3, 1024, 512);
  prep_w_kernel<<<(1024 * 3 * 1024 + 255) / 256, 256, 0, stream>>>(w4, WP4, 1024, 1024);
  prep_codebook_kernel<<<NC, 256, 0, stream>>>(cb, CBB, C2);

  // --- encoder (WMMA GEMM convs): grid = (ceil(T/128), C_out/64, B) ---
  conv_gemm_kernel<<<dim3(32, 8, B), 256, 0, stream>>>(
      X0, WP1, b1, H1, ZR, T1, 128, T1, 512, 1, 1);
  conv_gemm_kernel<<<dim3(16, 8, B), 256, 0, stream>>>(
      H1, WP2, b2, H2, ZR, T1, 512, T2, 512, 2, 1);
  conv_gemm_kernel<<<dim3(8, 16, B), 256, 0, stream>>>(
      H2, WP3, b3, H3, ZR, T2, 512, T3, 1024, 2, 1);
  conv_gemm_kernel<<<dim3(8, 16, B), 256, 0, stream>>>(
      H3, WP4, b4, H4, ZR, T3, 1024, T3, 1024, 1, 0);

  // --- quantize ---
  argmin_kernel<<<(B * T3) / 16, 256, 0, stream>>>(H4, CBB, C2, out, NC);
  lens_kernel<<<1, 32, 0, stream>>>(mel_lens, out, B * T3);
}
